// TransformerModel_84696755077832
// MI455X (gfx1250) — matrix operand order repaired and checked by reading, not compiled, not run on hardware
//
#include <hip/hip_runtime.h>
#include <hip/hip_bf16.h>
#include <math.h>

// model dims
#define D_MODEL 1024
#define NHEAD   16
#define HDIM    64      // D_MODEL / NHEAD
#define SEQ     512
#define BATCH   4
#define FFN     4096
#define VOCAB   32000
#define NLAYER  6
#define MROWS   (BATCH * SEQ)   // 2048
#define LN_EPS  1e-5f

typedef __attribute__((ext_vector_type(16))) __bf16          v16bf;
typedef __attribute__((ext_vector_type(8)))  float           v8f;
typedef __attribute__((ext_vector_type(8)))  unsigned short  us8;

union Frag { v16bf v; us8 h[2]; };
union U8   { us8 v; unsigned int w[4]; };

__device__ __forceinline__ unsigned short f2bf(float f) {
  // round-to-nearest-even fp32 -> bf16
  unsigned int x = __float_as_uint(f);
  x += 0x7FFFu + ((x >> 16) & 1u);
  return (unsigned short)(x >> 16);
}

__device__ __forceinline__ unsigned int pack2(float lo, float hi) {
  return (unsigned int)f2bf(lo) | ((unsigned int)f2bf(hi) << 16);
}

// convert 8 floats (two float4) -> 8 packed bf16 (16 bytes)
__device__ __forceinline__ us8 cvt_f4x2(float4 a, float4 b) {
  U8 u;
  u.w[0] = pack2(a.x, a.y);
  u.w[1] = pack2(a.z, a.w);
  u.w[2] = pack2(b.x, b.y);
  u.w[3] = pack2(b.z, b.w);
  return u.v;
}

__device__ __forceinline__ v8f wmma_bf16(v16bf a, v16bf b, v8f c) {
  return __builtin_amdgcn_wmma_f32_16x16x32_bf16(
      /*neg_a=*/false, a, /*neg_b=*/false, b,
      /*c_mod=*/(short)0, c, /*reuse_a=*/false, /*reuse_b=*/false);
}

// A fragment: 16 rows x 32 K, row-major LDS tile; K run hi*16+0..15 contiguous
__device__ __forceinline__ v16bf load_fragA(const unsigned short* rowp, int hi) {
  Frag f;
  f.h[0] = *(const us8*)(rowp + hi * 16);
  f.h[1] = *(const us8*)(rowp + hi * 16 + 8);
  return f.v;
}
// B fragment from transposed tile sBT[n][k]: K run hi*16+0..15 contiguous
__device__ __forceinline__ v16bf load_fragB(const unsigned short* colp, int hi) {
  Frag f;
  f.h[0] = *(const us8*)(colp + hi * 16);
  f.h[1] = *(const us8*)(colp + hi * 16 + 8);
  return f.v;
}

// ---------------------------------------------------------------------------
// Generic GEMM: C[M,N] = A[M,K]*Bw[K,N] + bias (+ReLU), bf16 WMMA, f32 accum.
// Block tile 128x128, BK=32, double-buffered LDS, 256 threads = 8 waves,
// wave tile 32x64 (8 WMMAs / 12 ds_load_b128 per K step).
// M%128==0, N%128==0, K%32==0 at all call sites.
// ---------------------------------------------------------------------------
template <bool RELU>
__global__ __launch_bounds__(256) void gemm_bias_kernel(
    const float* __restrict__ A, int lda,
    const float* __restrict__ Bw, int ldb,
    const float* __restrict__ bias,
    float* __restrict__ C, int ldc, int K) {
  __shared__ unsigned short sA[2][128][40];   // [buf][m][k], 80B rows
  __shared__ unsigned short sBT[2][128][40];  // [buf][n][k] transposed

  const int tid  = threadIdx.x;
  const int lane = tid & 31;
  const int wave = tid >> 5;
  const int wm   = wave & 3;   // 4 wave-rows of 32
  const int wn   = wave >> 2;  // 2 wave-cols of 64
  const int r    = lane & 15;
  const int hi   = lane >> 4;

  const int rowBase = blockIdx.y * 128;
  const int colBase = blockIdx.x * 128;

  // A fill: 128x32, 2 threads/row, 16 k each
  const int arow = tid >> 1;
  const int acol = (tid & 1) * 16;
  // B fill (transposing): 32 n-groups of 4 x 8 k-groups of 4
  const int bn = (tid & 31) * 4;   // n offset within tile (0..124)
  const int bk = (tid >> 5) * 4;   // k offset within tile (0..28)

  const float* aSrc = A + (size_t)(rowBase + arow) * lda + acol;
  const float* bSrc = Bw + colBase + bn;

#define FILL_TILE(buf, kk)                                                     \
  {                                                                            \
    const float4* ap = (const float4*)(aSrc + (kk));                           \
    *(us8*)&sA[buf][arow][acol]     = cvt_f4x2(ap[0], ap[1]);                  \
    *(us8*)&sA[buf][arow][acol + 8] = cvt_f4x2(ap[2], ap[3]);                  \
    const float4 q0 = *(const float4*)(bSrc + (size_t)((kk) + bk)     * ldb);  \
    const float4 q1 = *(const float4*)(bSrc + (size_t)((kk) + bk + 1) * ldb);  \
    const float4 q2 = *(const float4*)(bSrc + (size_t)((kk) + bk + 2) * ldb);  \
    const float4 q3 = *(const float4*)(bSrc + (size_t)((kk) + bk + 3) * ldb);  \
    uint2 t0, t1, t2, t3;                                                      \
    t0.x = pack2(q0.x, q1.x); t0.y = pack2(q2.x, q3.x);                        \
    t1.x = pack2(q0.y, q1.y); t1.y = pack2(q2.y, q3.y);                        \
    t2.x = pack2(q0.z, q1.z); t2.y = pack2(q2.z, q3.z);                        \
    t3.x = pack2(q0.w, q1.w); t3.y = pack2(q2.w, q3.w);                        \
    *(uint2*)&sBT[buf][bn + 0][bk] = t0;                                       \
    *(uint2*)&sBT[buf][bn + 1][bk] = t1;                                       \
    *(uint2*)&sBT[buf][bn + 2][bk] = t2;                                       \
    *(uint2*)&sBT[buf][bn + 3][bk] = t3;                                       \
  }

  v8f acc[2][4] = {};

  FILL_TILE(0, 0);
  __syncthreads();

  for (int k0 = 0; k0 < K; k0 += 32) {
    const int cur = (k0 >> 5) & 1;
    if (k0 + 32 < K) FILL_TILE(cur ^ 1, k0 + 32);
    if (k0 + 64 < K) {  // warm caches two tiles ahead (global_prefetch_b8)
      __builtin_prefetch(aSrc + k0 + 64, 0, 1);
      __builtin_prefetch(bSrc + (size_t)(k0 + 64 + bk) * ldb, 0, 1);
    }

    v16bf a0 = load_fragA(&sA[cur][wm * 32 + r][0], hi);
    v16bf a1 = load_fragA(&sA[cur][wm * 32 + 16 + r][0], hi);
    v16bf b0 = load_fragB(&sBT[cur][wn * 64 + r][0], hi);
    v16bf b1 = load_fragB(&sBT[cur][wn * 64 + 16 + r][0], hi);
    v16bf b2 = load_fragB(&sBT[cur][wn * 64 + 32 + r][0], hi);
    v16bf b3 = load_fragB(&sBT[cur][wn * 64 + 48 + r][0], hi);

    acc[0][0] = wmma_bf16(a0, b0, acc[0][0]);
    acc[0][1] = wmma_bf16(a0, b1, acc[0][1]);
    acc[0][2] = wmma_bf16(a0, b2, acc[0][2]);
    acc[0][3] = wmma_bf16(a0, b3, acc[0][3]);
    acc[1][0] = wmma_bf16(a1, b0, acc[1][0]);
    acc[1][1] = wmma_bf16(a1, b1, acc[1][1]);
    acc[1][2] = wmma_bf16(a1, b2, acc[1][2]);
    acc[1][3] = wmma_bf16(a1, b3, acc[1][3]);
    __syncthreads();
  }
#undef FILL_TILE

#pragma unroll
  for (int ti = 0; ti < 2; ++ti) {
#pragma unroll
    for (int tj = 0; tj < 4; ++tj) {
      const int col = colBase + wn * 64 + tj * 16 + r;
      const float bv = bias[col];
#pragma unroll
      for (int i = 0; i < 8; ++i) {
        const int row = rowBase + wm * 32 + ti * 16 + hi * 8 + i;
        float v = acc[ti][tj][i] + bv;
        if (RELU) v = fmaxf(v, 0.0f);
        C[(size_t)row * ldc + col] = v;
      }
    }
  }
}

// ---------------------------------------------------------------------------
// Attention scores: scores[bh,q,k] = (Q[b,q,h,:] . K[b,k,h,:]) / 8
// 64x64 score tile per block; K tile's row-major layout IS the transposed-B
// layout, so both fills are straight vector copies.
// ---------------------------------------------------------------------------
__global__ __launch_bounds__(256) void attn_scores_kernel(
    const float* __restrict__ Q, const float* __restrict__ Km,
    float* __restrict__ scores) {
  __shared__ unsigned short sQ[64][72];   // [m][d], 144B rows
  __shared__ unsigned short sKT[64][72];  // [n][d]

  const int tid  = threadIdx.x;
  const int lane = tid & 31;
  const int wave = tid >> 5;
  const int wm   = wave & 3;   // 4 wave-rows of 16
  const int wn   = wave >> 2;  // 2 wave-cols of 32
  const int r    = lane & 15;
  const int hi   = lane >> 4;

  const int bh = blockIdx.z;
  const int b  = bh / NHEAD;
  const int h  = bh % NHEAD;
  const int qBase = blockIdx.x * 64;
  const int kBase = blockIdx.y * 64;

  const float* qp = Q  + (size_t)b * SEQ * D_MODEL + (size_t)h * HDIM;
  const float* kp = Km + (size_t)b * SEQ * D_MODEL + (size_t)h * HDIM;

  const int lrow = tid >> 2;        // 0..63
  const int lcol = (tid & 3) * 16;  // 0,16,32,48
  {
    const float4* q4 = (const float4*)(qp + (size_t)(qBase + lrow) * D_MODEL + lcol);
    *(us8*)&sQ[lrow][lcol]     = cvt_f4x2(q4[0], q4[1]);
    *(us8*)&sQ[lrow][lcol + 8] = cvt_f4x2(q4[2], q4[3]);
    const float4* k4 = (const float4*)(kp + (size_t)(kBase + lrow) * D_MODEL + lcol);
    *(us8*)&sKT[lrow][lcol]     = cvt_f4x2(k4[0], k4[1]);
    *(us8*)&sKT[lrow][lcol + 8] = cvt_f4x2(k4[2], k4[3]);
  }
  __syncthreads();

  v8f acc[2] = {};
#pragma unroll
  for (int d0 = 0; d0 < HDIM; d0 += 32) {
    v16bf a  = load_fragA(&sQ[wm * 16 + r][d0], hi);
    v16bf b0 = load_fragB(&sKT[wn * 32 + r][d0], hi);
    v16bf b1 = load_fragB(&sKT[wn * 32 + 16 + r][d0], hi);
    acc[0] = wmma_bf16(a, b0, acc[0]);
    acc[1] = wmma_bf16(a, b1, acc[1]);
  }

  float* out = scores + (size_t)bh * SEQ * SEQ;
#pragma unroll
  for (int tj = 0; tj < 2; ++tj) {
    const int col = kBase + wn * 32 + tj * 16 + r;
#pragma unroll
    for (int i = 0; i < 8; ++i) {
      const int row = qBase + wm * 16 + hi * 8 + i;
      out[(size_t)row * SEQ + col] = acc[tj][i] * 0.125f;  // 1/sqrt(64)
    }
  }
}

// ---------------------------------------------------------------------------
// Softmax over keys with causal + pad mask; handles NaN and all(-inf) rows.
// ---------------------------------------------------------------------------
__global__ __launch_bounds__(256) void softmax_kernel(
    float* __restrict__ scores, const float* __restrict__ causal,
    const int* __restrict__ ids) {
  __shared__ float red[256];
  const int rowid = blockIdx.x;       // bh*SEQ + q
  const int bh = rowid / SEQ;
  const int q  = rowid % SEQ;
  const int b  = bh / NHEAD;
  float* row = scores + (size_t)rowid * SEQ;
  const int t = threadIdx.x;

  float s[2];
  float mx = -INFINITY;
#pragma unroll
  for (int j = 0; j < 2; ++j) {
    const int k = t + j * 256;
    float v = row[k] + causal[q * SEQ + k] +
              (ids[b * SEQ + k] == 0 ? -INFINITY : 0.0f);
    if (v != v) v = 0.0f;  // NaN -> 0 (matches reference)
    s[j] = v;
    mx = fmaxf(mx, v);
  }
  red[t] = mx; __syncthreads();
  for (int off = 128; off > 0; off >>= 1) {
    if (t < off) red[t] = fmaxf(red[t], red[t + off]);
    __syncthreads();
  }
  mx = red[0]; __syncthreads();

  if (isinf(mx) && mx < 0.0f) {  // entire row masked -> uniform softmax
    s[0] = 0.0f; s[1] = 0.0f; mx = 0.0f;
  }
  float sum = 0.0f;
#pragma unroll
  for (int j = 0; j < 2; ++j) { s[j] = __expf(s[j] - mx); sum += s[j]; }
  red[t] = sum; __syncthreads();
  for (int off = 128; off > 0; off >>= 1) {
    if (t < off) red[t] += red[t + off];
    __syncthreads();
  }
  const float inv = 1.0f / red[0];
#pragma unroll
  for (int j = 0; j < 2; ++j) row[t + j * 256] = s[j] * inv;
}

// ---------------------------------------------------------------------------
// Context: ctx[b,q,h,:] = sum_k W[bh,q,k] * V[b,k,h,:]  (M=512,N=64,K=512)
// Double-buffered LDS pipeline over K.
// ---------------------------------------------------------------------------
__global__ __launch_bounds__(256) void attn_ctx_kernel(
    const float* __restrict__ W, const float* __restrict__ Vm,
    float* __restrict__ ctx) {
  __shared__ unsigned short sW[2][64][40];   // [buf][m][k]
  __shared__ unsigned short sVT[2][64][40];  // [buf][n][k] transposed

  const int tid  = threadIdx.x;
  const int lane = tid & 31;
  const int wave = tid >> 5;
  const int wm   = wave & 3;   // 4 wave-rows of 16
  const int wn   = wave >> 2;  // 2 wave-cols of 32
  const int r    = lane & 15;
  const int hi   = lane >> 4;

  const int bh = blockIdx.z;
  const int b  = bh / NHEAD;
  const int h  = bh % NHEAD;
  const int mBase = blockIdx.x * 64;

  const float* wp = W + (size_t)bh * SEQ * SEQ;
  const float* vp = Vm + (size_t)b * SEQ * D_MODEL + (size_t)h * HDIM;

  const int wrow = tid >> 2;        // 0..63
  const int wcol = (tid & 3) * 8;   // 0,8,16,24
  const int vn   = (tid & 15) * 4;  // n offset
  const int vk   = (tid >> 4) * 2;  // k offset

#define FILL_CTX(buf, kk)                                                      \
  {                                                                            \
    const float4* w4 =                                                         \
        (const float4*)(wp + (size_t)(mBase + wrow) * SEQ + (kk) + wcol);      \
    *(us8*)&sW[buf][wrow][wcol] = cvt_f4x2(w4[0], w4[1]);                      \
    const float4 r0 = *(const float4*)(vp + (size_t)((kk) + vk)     * D_MODEL + vn); \
    const float4 r1 = *(const float4*)(vp + (size_t)((kk) + vk + 1) * D_MODEL + vn); \
    *(unsigned int*)&sVT[buf][vn + 0][vk] = pack2(r0.x, r1.x);                 \
    *(unsigned int*)&sVT[buf][vn + 1][vk] = pack2(r0.y, r1.y);                 \
    *(unsigned int*)&sVT[buf][vn + 2][vk] = pack2(r0.z, r1.z);                 \
    *(unsigned int*)&sVT[buf][vn + 3][vk] = pack2(r0.w, r1.w);                 \
  }

  v8f acc[2] = {};

  FILL_CTX(0, 0);
  __syncthreads();

  for (int k0 = 0; k0 < SEQ; k0 += 32) {
    const int cur = (k0 >> 5) & 1;
    if (k0 + 32 < SEQ) FILL_CTX(cur ^ 1, k0 + 32);

    v16bf a  = load_fragA(&sW[cur][wm * 16 + r][0], hi);
    v16bf b0 = load_fragB(&sVT[cur][wn * 32 + r][0], hi);
    v16bf b1 = load_fragB(&sVT[cur][wn * 32 + 16 + r][0], hi);
    acc[0] = wmma_bf16(a, b0, acc[0]);
    acc[1] = wmma_bf16(a, b1, acc[1]);
    __syncthreads();
  }
#undef FILL_CTX

#pragma unroll
  for (int tj = 0; tj < 2; ++tj) {
    const int col = wn * 32 + tj * 16 + r;  // 0..63 within head
#pragma unroll
    for (int i = 0; i < 8; ++i) {
      const int row = mBase + wm * 16 + hi * 8 + i;
      ctx[(size_t)b * SEQ * D_MODEL + (size_t)row * D_MODEL + h * HDIM + col] =
          acc[tj][i];
    }
  }
}

// ---------------------------------------------------------------------------
// Embedding * sqrt(D) + sinusoidal positional encoding.
// ---------------------------------------------------------------------------
__global__ __launch_bounds__(256) void embed_pe_kernel(
    const int* __restrict__ ids, const float* __restrict__ emb,
    float* __restrict__ x) {
  const int rowid = blockIdx.x;  // b*SEQ + s
  const int s  = rowid % SEQ;
  const int id = ids[rowid];
  const int t  = threadIdx.x;
#pragma unroll
  for (int j = 0; j < 4; ++j) {
    const int d  = t + j * 256;
    const float i2  = (float)((d >> 1) * 2);
    const float div = __expf(i2 * (-9.210340371976184f / (float)D_MODEL)); // -ln(1e4)/D
    const float ang = (float)s * div;
    const float pe  = (d & 1) ? __cosf(ang) : __sinf(ang);
    x[(size_t)rowid * D_MODEL + d] = emb[(size_t)id * D_MODEL + d] * 32.0f + pe;
  }
}

// ---------------------------------------------------------------------------
// x = LayerNorm(x + y) * g + b   (in place on x), one block per row of 1024.
// ---------------------------------------------------------------------------
__global__ __launch_bounds__(256) void add_ln_kernel(
    float* __restrict__ x, const float* __restrict__ y,
    const float* __restrict__ g, const float* __restrict__ bn) {
  __shared__ float red[256];
  const int rowid = blockIdx.x;
  float* xr = x + (size_t)rowid * D_MODEL;
  const float* yr = y + (size_t)rowid * D_MODEL;
  const int t = threadIdx.x;

  float v[4];
  float sum = 0.0f;
#pragma unroll
  for (int j = 0; j < 4; ++j) {
    v[j] = xr[t + j * 256] + yr[t + j * 256];
    sum += v[j];
  }
  red[t] = sum; __syncthreads();
  for (int off = 128; off > 0; off >>= 1) {
    if (t < off) red[t] += red[t + off];
    __syncthreads();
  }
  const float mean = red[0] * (1.0f / D_MODEL);
  __syncthreads();

  float sq = 0.0f;
#pragma unroll
  for (int j = 0; j < 4; ++j) {
    const float d = v[j] - mean;
    sq += d * d;
  }
  red[t] = sq; __syncthreads();
  for (int off = 128; off > 0; off >>= 1) {
    if (t < off) red[t] += red[t + off];
    __syncthreads();
  }
  const float rstd = rsqrtf(red[0] * (1.0f / D_MODEL) + LN_EPS);

#pragma unroll
  for (int j = 0; j < 4; ++j) {
    const int c = t + j * 256;
    xr[c] = (v[j] - mean) * rstd * g[c] + bn[c];
  }
}

// ---------------------------------------------------------------------------
extern "C" void kernel_launch(void* const* d_in, const int* in_sizes, int n_in,
                              void* d_out, int out_size, void* d_ws, size_t ws_size,
                              hipStream_t stream) {
  (void)in_sizes; (void)n_in; (void)out_size; (void)ws_size;

  const int*   ids    = (const int*)d_in[0];
  const float* causal = (const float*)d_in[1];
  const float* emb    = (const float*)d_in[2];
  const float* Wq  = (const float*)d_in[3];
  const float* bq  = (const float*)d_in[4];
  const float* Wk  = (const float*)d_in[5];
  const float* bk  = (const float*)d_in[6];
  const float* Wv  = (const float*)d_in[7];
  const float* bv  = (const float*)d_in[8];
  const float* Wo  = (const float*)d_in[9];
  const float* bo  = (const float*)d_in[10];
  const float* g1  = (const float*)d_in[11];
  const float* b1n = (const float*)d_in[12];
  const float* W1  = (const float*)d_in[13];
  const float* b1f = (const float*)d_in[14];
  const float* W2  = (const float*)d_in[15];
  const float* b2f = (const float*)d_in[16];
  const float* g2  = (const float*)d_in[17];
  const float* b2n = (const float*)d_in[18];
  const float* Wout = (const float*)d_in[19];
  const float* bout = (const float*)d_in[20];

  float* ws = (float*)d_ws;
  size_t off = 0;
  float* x      = ws + off; off += (size_t)MROWS * D_MODEL;
  float* q      = ws + off; off += (size_t)MROWS * D_MODEL;
  float* kbuf   = ws + off; off += (size_t)MROWS * D_MODEL;
  float* vbuf   = ws + off; off += (size_t)MROWS * D_MODEL;
  float* ctxb   = ws + off; off += (size_t)MROWS * D_MODEL;
  float* tmp    = ws + off; off += (size_t)MROWS * D_MODEL;
  float* ffh    = ws + off; off += (size_t)MROWS * FFN;
  float* scores = ws + off; off += (size_t)BATCH * NHEAD * SEQ * SEQ;

  embed_pe_kernel<<<MROWS, 256, 0, stream>>>(ids, emb, x);

  const dim3 gDD(D_MODEL / 128, MROWS / 128);
  const dim3 gDF(FFN / 128, MROWS / 128);
  const dim3 gAttnS(SEQ / 64, SEQ / 64, BATCH * NHEAD);
  const dim3 gAttnC(SEQ / 64, 1, BATCH * NHEAD);

  for (int l = 0; l < NLAYER; ++l) {
    const size_t oDD = (size_t)l * D_MODEL * D_MODEL;
    const size_t oD  = (size_t)l * D_MODEL;
    const size_t oDF = (size_t)l * D_MODEL * FFN;
    const size_t oF  = (size_t)l * FFN;

    gemm_bias_kernel<false><<<gDD, 256, 0, stream>>>(x, D_MODEL, Wq + oDD, D_MODEL, bq + oD, q,    D_MODEL, D_MODEL);
    gemm_bias_kernel<false><<<gDD, 256, 0, stream>>>(x, D_MODEL, Wk + oDD, D_MODEL, bk + oD, kbuf, D_MODEL, D_MODEL);
    gemm_bias_kernel<false><<<gDD, 256, 0, stream>>>(x, D_MODEL, Wv + oDD, D_MODEL, bv + oD, vbuf, D_MODEL, D_MODEL);

    attn_scores_kernel<<<gAttnS, 256, 0, stream>>>(q, kbuf, scores);
    softmax_kernel<<<BATCH * NHEAD * SEQ, 256, 0, stream>>>(scores, causal, ids);
    attn_ctx_kernel<<<gAttnC, 256, 0, stream>>>(scores, vbuf, ctxb);

    gemm_bias_kernel<false><<<gDD, 256, 0, stream>>>(ctxb, D_MODEL, Wo + oDD, D_MODEL, bo + oD, tmp, D_MODEL, D_MODEL);
    add_ln_kernel<<<MROWS, 256, 0, stream>>>(x, tmp, g1 + oD, b1n + oD);

    gemm_bias_kernel<true ><<<gDF, 256, 0, stream>>>(x,   D_MODEL, W1 + oDF, FFN,     b1f + oF, ffh, FFN,     D_MODEL);
    gemm_bias_kernel<false><<<gDD, 256, 0, stream>>>(ffh, FFN,     W2 + oDF, D_MODEL, b2f + oD, tmp, D_MODEL, FFN);
    add_ln_kernel<<<MROWS, 256, 0, stream>>>(x, tmp, g2 + oD, b2n + oD);
  }

  gemm_bias_kernel<false><<<dim3(VOCAB / 128, MROWS / 128), 256, 0, stream>>>(
      x, D_MODEL, Wout, VOCAB, bout, (float*)d_out, VOCAB, D_MODEL);
}
